// Spatom_39840116638199
// MI455X (gfx1250) — compile-verified, matrix operand order checked by reading, not run on hardware
//
#include <hip/hip_runtime.h>
#include <hip/hip_bf16.h>

// ---------------------------------------------------------------------------
// CDNA5 (gfx1250, wave32) implementation.
// GEMMs run on v_wmma_f32_16x16x32_bf16 with a 2-deep software pipeline:
// next k-step's A (fp32, packed-cvt to bf16) and B (pre-transposed bf16)
// tiles are issued before the current step's WMMAs, so loads overlap the
// matrix pipe and no WAR hazard NOPs are needed.
// ---------------------------------------------------------------------------

typedef __attribute__((ext_vector_type(16))) __bf16          v16bf;
typedef __attribute__((ext_vector_type(8)))  __bf16          v8bf;
typedef __attribute__((ext_vector_type(8)))  float           v8f;
typedef __attribute__((ext_vector_type(4)))  float           v4f;

// ---------------------------------------------------------------------------
// Weight convert + transpose: dst[n*K + k] = bf16(src[k*ld + n])
// ---------------------------------------------------------------------------
__global__ void convT_kernel(const float* __restrict__ src,
                             __bf16* __restrict__ dst,
                             int K, int NC, int ld) {
    int t = blockIdx.x * blockDim.x + threadIdx.x;
    if (t >= K * NC) return;
    int n = t / K, k = t % K;
    dst[t] = (__bf16)src[(size_t)k * ld + n];
}

// ---------------------------------------------------------------------------
// WMMA GEMM: out[N x 64*colChunks] = epi(A[N x Kdim] @ Wt^T + bias)
// Wt is bf16, stored transposed: Wt[col][Kdim].  Kdim % 64 == 0 required.
// One wave = 16 rows x 64 cols (4 WMMA col tiles). wave32 layouts:
//   A 16x32 bf16 : lane = M%16, lane-half selects K sub-blocks {0..7,16..23}
//                  vs {8..15,24..31}; vector elems 0..7 / 8..15 contiguous K.
//   B 32x16 bf16 : lane = N%16, same K enumeration.
//   C/D 16x16 f32: VGPR r -> M = r + 8*(lane/16), N = lane%16.
// EPI: 0 none(+bias) | 1 leaky0.2 | 2 relu | 3 lin+support | 4 chen-combine
// ---------------------------------------------------------------------------
struct KTile {
    v4f  a0, a1, a2, a3;     // 16 fp32 A elements (two contiguous K-chunks)
    v8bf b[4][2];            // 4 col tiles x two contiguous K-chunks
};

template<int EPI>
__global__ __launch_bounds__(256) void gemm_bf16_wmma(
    const float* __restrict__ A, int lda,
    const __bf16* __restrict__ Wt, int Kdim,
    const float* __restrict__ bias,
    float* __restrict__ outF, float* __restrict__ outF2, int ldo,
    const float* __restrict__ e0, const float* __restrict__ e1,
    float c0, float c1, float c2,
    int rowTiles, int colChunks)
{
    int wave  = blockIdx.x * blockDim.y + threadIdx.y;
    int tile  = wave / colChunks;
    int chunk = wave - tile * colChunks;
    if (tile >= rowTiles) return;           // wave-uniform: EXEC stays all-1s

    const int lane = threadIdx.x;
    const int lr   = lane & 15;
    const int half = lane >> 4;
    const int r0   = tile * 16;
    const int cb   = chunk * 64;

    v8f acc[4];
    #pragma unroll
    for (int t = 0; t < 4; ++t)
        #pragma unroll
        for (int r = 0; r < 8; ++r) acc[t][r] = 0.0f;

    const float* arow = A + (size_t)(r0 + lr) * lda + half * 8;
    const __bf16* brow[4];
    #pragma unroll
    for (int t = 0; t < 4; ++t)
        brow[t] = Wt + (size_t)(cb + t * 16 + lr) * Kdim + half * 8;

    auto loadT = [&](int k0, KTile& T) {
        T.a0 = *(const v4f*)(arow + k0);
        T.a1 = *(const v4f*)(arow + k0 + 4);
        T.a2 = *(const v4f*)(arow + k0 + 16);
        T.a3 = *(const v4f*)(arow + k0 + 20);
        #pragma unroll
        for (int t = 0; t < 4; ++t) {
            T.b[t][0] = *(const v8bf*)(brow[t] + k0);
            T.b[t][1] = *(const v8bf*)(brow[t] + k0 + 16);
        }
    };
    auto computeT = [&](const KTile& T) {
        v16bf av;
        #pragma unroll
        for (int i = 0; i < 4; ++i) {
            av[i]      = (__bf16)T.a0[i];
            av[4 + i]  = (__bf16)T.a1[i];
            av[8 + i]  = (__bf16)T.a2[i];
            av[12 + i] = (__bf16)T.a3[i];
        }
        #pragma unroll
        for (int t = 0; t < 4; ++t) {
            v16bf bv;
            #pragma unroll
            for (int i = 0; i < 8; ++i) {
                bv[i]     = T.b[t][0][i];
                bv[8 + i] = T.b[t][1][i];
            }
            acc[t] = __builtin_amdgcn_wmma_f32_16x16x32_bf16(
                false, av, false, bv, (short)0, acc[t], false, false);
        }
    };

    // ---- 2-deep software-pipelined main loop (Kdim % 64 == 0)
    KTile T0, T1;
    loadT(0, T0);
    for (int k0 = 0; k0 < Kdim; k0 += 64) {
        loadT(k0 + 32, T1);                 // prefetch while T0 computes
        computeT(T0);
        if (k0 + 64 < Kdim) loadT(k0 + 64, T0);
        computeT(T1);
    }

    // ---- epilogue + store (coalesced per 16-lane half)
    #pragma unroll
    for (int t = 0; t < 4; ++t) {
        int col = cb + t * 16 + lr;
        #pragma unroll
        for (int r = 0; r < 8; ++r) {
            int row   = r0 + half * 8 + r;
            size_t oi = (size_t)row * ldo + col;
            float v   = acc[t][r];
            if (EPI == 0) {
                if (bias) v += bias[col];
                outF[oi] = v;
            } else if (EPI == 1) {
                v += bias[col];
                outF[oi] = v > 0.f ? v : 0.2f * v;
            } else if (EPI == 2) {
                v += bias[col];
                outF[oi] = v > 0.f ? v : 0.f;
            } else if (EPI == 3) {          // lin_out + support
                v += bias[col];
                outF[oi]  = v;
                outF2[oi] = c1 * v + c2 * e0[oi];
            } else {                         // chen combine
                float x = c0 * v + c1 * e0[oi] + c2 * e1[oi];
                outF[oi] = x > 0.f ? x : 0.f;
            }
        }
    }
}

// ---------------------------------------------------------------------------
// Per-edge window + 3 coords (only first 3 cols of conv_w2 matter).
// aw[e][c] = window * coord_c ; win[e] = window
// ---------------------------------------------------------------------------
__global__ void window_kernel(const float* __restrict__ dij,
                              const float* __restrict__ xyz,
                              const float* __restrict__ w1,
                              const float* __restrict__ b1,
                              const float* __restrict__ w2,
                              const float* __restrict__ b2,
                              float* __restrict__ win,
                              float* __restrict__ aw, int NK)
{
    int t = blockIdx.x * blockDim.x + threadIdx.x;
    if (t >= NK) return;
    const float* d = dij + (size_t)t * 3;
    float d2 = d[0] * d[0] + d[1] * d[1] + d[2] * d[2];
    float w  = d2 < 169.0f ? expf(-d2 * (1.0f / 338.0f)) : 0.0f;

    const float* p = xyz + (size_t)t * 3;
    float c[3] = {b2[0], b2[1], b2[2]};
    #pragma unroll
    for (int u = 0; u < 16; ++u) {
        float q = p[0] * w1[u] + p[1] * w1[16 + u] + p[2] * w1[32 + u] + b1[u];
        q = q > 0.f ? q : 0.f;
        c[0] += q * w2[u * 64 + 0];
        c[1] += q * w2[u * 64 + 1];
        c[2] += q * w2[u * 64 + 2];
    }
    win[t] = w;
    aw[(size_t)t * 3 + 0] = w * c[0];
    aw[(size_t)t * 3 + 1] = w * c[1];
    aw[(size_t)t * 3 + 2] = w * c[2];
}

// ---------------------------------------------------------------------------
// F[n,h] = sum_k sum_c aw[n,k,c] * mg[idx[n,k], c*64+h] + wsum*mlp_b[h]
// then batch-norm -> Fn. 64 threads / node; mg rows are L2-resident (15 MB).
// ---------------------------------------------------------------------------
__global__ __launch_bounds__(256) void gather_kernel(
    const int* __restrict__ idx, const float* __restrict__ win,
    const float* __restrict__ aw, const float* __restrict__ mg,
    const float* __restrict__ mlp_b, const float* __restrict__ bn_g,
    const float* __restrict__ bn_b, float* __restrict__ Fn, int Nn)
{
    int t = blockIdx.x * blockDim.x + threadIdx.x;
    int n = t >> 6, h = t & 63;
    if (n >= Nn) return;
    float acc = 0.f, wsum = 0.f;
    #pragma unroll
    for (int k = 0; k < 16; ++k) {
        int e = n * 16 + k;
        int j = idx[e];
        wsum += win[e];
        const float* row = mg + (size_t)j * 192;
        const float* a   = aw + (size_t)e * 3;
        acc += a[0] * row[h] + a[1] * row[64 + h] + a[2] * row[128 + h];
    }
    acc += wsum * mlp_b[h];
    const float inv = 0.99999500003750f;   // 1/sqrt(1+1e-5)
    Fn[(size_t)n * 64 + h] = acc * inv * bn_g[h] + bn_b[h];
}

// ---------------------------------------------------------------------------
// Head: sigmoid(leaky(leaky(h@w1+b1)*w2+b2))
// ---------------------------------------------------------------------------
__global__ void out_kernel(const float* __restrict__ h,
                           const float* __restrict__ w1,
                           const float* __restrict__ b1,
                           const float* __restrict__ w2,
                           const float* __restrict__ b2,
                           float* __restrict__ out, int Nn)
{
    int n = blockIdx.x * blockDim.x + threadIdx.x;
    if (n >= Nn) return;
    const float* r = h + (size_t)n * 64;
    float s = b1[0];
    #pragma unroll
    for (int c = 0; c < 64; ++c) s += r[c] * w1[c];
    s = s > 0.f ? s : 0.2f * s;
    s = s * w2[0] + b2[0];
    s = s > 0.f ? s : 0.2f * s;
    out[n] = 1.0f / (1.0f + expf(-s));
}

// ---------------------------------------------------------------------------
extern "C" void kernel_launch(void* const* d_in, const int* in_sizes, int n_in,
                              void* d_out, int out_size, void* d_ws, size_t ws_size,
                              hipStream_t stream)
{
    (void)in_sizes; (void)n_in; (void)out_size; (void)ws_size;
    const int N = 20000, K = 16, DIN = 1024, H = 64;

    const float* x       = (const float*)d_in[0];
    const int*   nbr_idx = (const int*)  d_in[1];
    const float* nbr_xyz = (const float*)d_in[2];
    const float* nbr_dij = (const float*)d_in[3];
    auto P = [&](int i) { return (const float*)d_in[i]; };
    // param indices: 4 fc_w,5 fc_b,6 in0_w1,7 in0_b1,8 in0_w2,9 in0_b2,
    // 10..17 in{1,2}_{w1,b1,w2,b2}, per-j block base=18+11j:
    //   +0 conv_w1,+1 conv_b1,+2 conv_w2,+3 conv_b2,+4 bn_g,+5 bn_b,
    //   +6 ll_w1,+7 ll_b1,+8 ll_w2,+9 ll_b2,+10 chen_w
    // 51 mlp_w,52 mlp_b,53 out_w1,54 out_b1,55 out_w2,56 out_b2

    // ---- workspace bump allocator
    char* ws = (char*)d_ws;
    size_t off = 0;
    auto alloc = [&](size_t bytes) -> void* {
        void* p = ws + off; off += (bytes + 255) & ~(size_t)255; return p;
    };
    float* resF = (float*)alloc((size_t)N * H * 4);
    float* t1F  = (float*)alloc((size_t)N * H * 4);
    float* gF   = (float*)alloc((size_t)N * H * 4);
    float* mgF  = (float*)alloc((size_t)N * 192 * 4);
    float* FnF  = (float*)alloc((size_t)N * H * 4);
    float* linF = (float*)alloc((size_t)N * H * 4);
    float* supF = (float*)alloc((size_t)N * H * 4);
    float* hF   = (float*)alloc((size_t)N * H * 4);
    float* win  = (float*)alloc((size_t)N * K * 4);
    float* aw   = (float*)alloc((size_t)N * K * 3 * 4);
    __bf16* fcT   = (__bf16*)alloc((size_t)64 * 1024 * 2);
    __bf16* i0w1T = (__bf16*)alloc((size_t)64 * 1024 * 2);
    __bf16* mgT   = (__bf16*)alloc((size_t)192 * 64 * 2);
    __bf16* i0w2T = (__bf16*)alloc(4096 * 2);
    __bf16* iw1T2[2]; __bf16* iw2T2[2];
    for (int j = 0; j < 2; ++j) {
        iw1T2[j] = (__bf16*)alloc(4096 * 2);
        iw2T2[j] = (__bf16*)alloc(4096 * 2);
    }
    __bf16 *llw1T[3], *llw2T[3], *chenT[3];
    for (int j = 0; j < 3; ++j) {
        llw1T[j] = (__bf16*)alloc(4096 * 2);
        llw2T[j] = (__bf16*)alloc(4096 * 2);
        chenT[j] = (__bf16*)alloc(4096 * 2);
    }

    // ---- weight prep (convert + transpose once per launch; deterministic)
    auto cvt = [&](const float* src, __bf16* dst, int Kd, int NC, int ld) {
        int n = Kd * NC;
        convT_kernel<<<(n + 255) / 256, 256, 0, stream>>>(src, dst, Kd, NC, ld);
    };
    cvt(P(4), fcT, 1024, 64, 64);
    cvt(P(6), i0w1T, 1024, 64, 64);
    cvt(P(8), i0w2T, 64, 64, 64);
    cvt(P(10), iw1T2[0], 64, 64, 64);
    cvt(P(12), iw2T2[0], 64, 64, 64);
    cvt(P(14), iw1T2[1], 64, 64, 64);
    cvt(P(16), iw2T2[1], 64, 64, 64);
    for (int j = 0; j < 3; ++j) {
        int base = 18 + 11 * j;
        cvt(P(base + 6), llw1T[j], 64, 64, 64);
        cvt(P(base + 8), llw2T[j], 64, 64, 64);
        cvt(P(base + 10), chenT[j], 64, 64, 64);
    }
    for (int c = 0; c < 3; ++c)                       // mlp_w -> [W0|W1|W2]^T
        cvt(P(51) + (size_t)c * 64 * 64, mgT + (size_t)c * 64 * 64, 64, 64, 64);

    // ---- GEMM dispatcher
    const int rowTiles = N / 16;                      // 1250
    auto gemm = [&](int epi, const float* A, int lda, const __bf16* Wt,
                    int Kd, const float* bias, float* oF, float* oF2, int ldo,
                    const float* e0, const float* e1,
                    float c0, float c1, float c2, int colChunks) {
        dim3 blk(32, 8);
        int waves = rowTiles * colChunks;
        int grid  = (waves + 7) / 8;
        switch (epi) {
        case 0: gemm_bf16_wmma<0><<<grid, blk, 0, stream>>>(A, lda, Wt, Kd, bias, oF, oF2, ldo, e0, e1, c0, c1, c2, rowTiles, colChunks); break;
        case 1: gemm_bf16_wmma<1><<<grid, blk, 0, stream>>>(A, lda, Wt, Kd, bias, oF, oF2, ldo, e0, e1, c0, c1, c2, rowTiles, colChunks); break;
        case 2: gemm_bf16_wmma<2><<<grid, blk, 0, stream>>>(A, lda, Wt, Kd, bias, oF, oF2, ldo, e0, e1, c0, c1, c2, rowTiles, colChunks); break;
        case 3: gemm_bf16_wmma<3><<<grid, blk, 0, stream>>>(A, lda, Wt, Kd, bias, oF, oF2, ldo, e0, e1, c0, c1, c2, rowTiles, colChunks); break;
        default: gemm_bf16_wmma<4><<<grid, blk, 0, stream>>>(A, lda, Wt, Kd, bias, oF, oF2, ldo, e0, e1, c0, c1, c2, rowTiles, colChunks); break;
        }
    };

    // residual = x @ fc_w + fc_b
    gemm(0, x, DIN, fcT, DIN, P(5), resF, nullptr, 64, nullptr, nullptr, 0, 0, 0, 1);

    const float thetas[3] = {0.91629073f, 0.55961579f, 0.40546511f};

    for (int j = 0; j < 3; ++j) {
        const float* Ain = (j == 0) ? x : hF;
        int Kin = (j == 0) ? DIN : H;
        const __bf16* w1T = (j == 0) ? i0w1T : iw1T2[j - 1];
        const __bf16* w2T = (j == 0) ? i0w2T : iw2T2[j - 1];
        const float* b1 = P(j == 0 ? 7 : (j == 1 ? 11 : 15));
        const float* b2 = P(j == 0 ? 9 : (j == 1 ? 13 : 17));
        int base = 18 + 11 * j;

        // g = leaky(leaky(h @ w1 + b1) @ w2 + b2)   (per-node; gather commutes)
        gemm(1, Ain, Kin, w1T, Kin, b1, t1F, nullptr, 64, nullptr, nullptr, 0, 0, 0, 1);
        gemm(1, t1F, 64, w2T, 64, b2, gF, nullptr, 64, nullptr, nullptr, 0, 0, 0, 1);

        // mg = g @ [W0|W1|W2]   (N x 192)
        gemm(0, gF, 64, mgT, 64, nullptr, mgF, nullptr, 192, nullptr, nullptr, 0, 0, 0, 3);

        // per-edge window * coords
        window_kernel<<<(N * K + 255) / 256, 256, 0, stream>>>(
            nbr_dij, nbr_xyz, P(base), P(base + 1), P(base + 2), P(base + 3),
            win, aw, N * K);

        // F -> Fn
        gather_kernel<<<(N * 64 + 255) / 256, 256, 0, stream>>>(
            nbr_idx, win, aw, mgF, P(52), P(base + 4), P(base + 5), FnF, N);

        // lin_out = relu(Fn @ ll_w1 + b1) @ ll_w2 + b2 ; support = .3*lin+.7*res
        gemm(2, FnF, 64, llw1T[j], 64, P(base + 7), t1F, nullptr, 64, nullptr, nullptr, 0, 0, 0, 1);
        gemm(3, t1F, 64, llw2T[j], 64, P(base + 9), linF, supF, 64, resF, nullptr,
             0.f, 0.3f, 0.7f, 1);

        // h = relu(theta*(sup @ chen) + (1-theta)*sup + lin)
        gemm(4, supF, 64, chenT[j], 64, nullptr, hF, nullptr, 64, supF, linF,
             thetas[j], 1.0f - thetas[j], 1.0f, 1);
    }

    out_kernel<<<(N + 255) / 256, 256, 0, stream>>>(
        hF, P(53), P(54), P(55), P(56), (float*)d_out, N);
}
